// MLADecode_47682726920485
// MI455X (gfx1250) — matrix operand order repaired and checked by reading, not compiled, not run on hardware
//
#include <hip/hip_runtime.h>
#include <hip/hip_bf16.h>

// MLA decode (flash-decode, split-S) for MI455X / gfx1250.
// Memory-bound problem (~305 MB min traffic @ 23.3 TB/s ~= 13us), so:
//  - K/K_pe streamed from HBM exactly once, reused from LDS for both GEMMs
//  - 256 workgroups (32 batches x 8 S-splits) to saturate HBM
//  - all matrix math via v_wmma_f32_16x16x32_bf16 (f32 accumulate, f32 softmax)

typedef __attribute__((ext_vector_type(16))) __bf16 v16bf;
typedef __attribute__((ext_vector_type(8)))  float  v8f;

namespace {
constexpr int kB  = 32;    // batches
constexpr int kH  = 128;   // query heads
constexpr int kS  = 4096;  // sequence (keys)
constexpr int kD  = 512;   // latent dim (nope), also V dim
constexpr int kP  = 64;    // rope dim
constexpr int kDT = 576;   // kD + kP
constexpr int kSplits  = 8;
constexpr int kChunk   = kS / kSplits;  // 512 keys per split
constexpr int kTile    = 32;            // keys per inner tile
constexpr int kNT      = kChunk / kTile;
constexpr int kThreads = 512;           // 16 wave32
}

struct B32x8 { uint4 lo, hi; };

// A-matrix 16x32 bf16 fragment (contiguous-half layout):
// lanes 0-15 hold K=0..15 of row M=lane; lanes 16-31 hold K=16..31.
static __device__ __forceinline__ v16bf frag_ilv(const __bf16* rowbase, bool lolane) {
  const char* p = (const char*)rowbase + (lolane ? 0 : 32);
  B32x8 r;
  r.lo = *(const uint4*)(p);
  r.hi = *(const uint4*)(p + 16);
  return __builtin_bit_cast(v16bf, r);
}

// B-matrix 32x16 bf16 fragment (contiguous-half convention): lane holds 16
// consecutive K values of its column; caller passes pointer to first element.
static __device__ __forceinline__ v16bf frag_row(const __bf16* p) {
  B32x8 r;
  r.lo = *(const uint4*)(p);
  r.hi = *(const uint4*)(p + 8);
  return __builtin_bit_cast(v16bf, r);
}

__global__ __launch_bounds__(kThreads, 1) void mla_split(
    const float* __restrict__ Q, const float* __restrict__ Qpe,
    const float* __restrict__ K, const float* __restrict__ Kpe,
    float* __restrict__ po, float* __restrict__ pm, float* __restrict__ pl)
{
  __shared__ __bf16 sQ[kH * kDT];       // 144 KB: Q||Q_pe bf16
  __shared__ __bf16 sK[kTile * kDT];    //  36 KB: K tile row-major (GEMM1 B)
  __shared__ __bf16 sKt[kDT * kTile];   //  36 KB: K tile transposed (GEMM2 B)
  __shared__ float  sS[kH * kTile];     //  16 KB: scaled scores
  __shared__ __bf16 sP[kH * kTile];     //   8 KB: probs bf16 (GEMM2 A)
  __shared__ float  sM[kH], sL[kH], sA[kH];

  const int tid    = threadIdx.x;
  const int b      = blockIdx.y;
  const int split  = blockIdx.x;
  const int wv     = tid >> 5;
  const int lane   = tid & 31;
  const int col    = lane & 15;
  const bool lolane = lane < 16;
  const int gm = wv & 7;   // g-tile (rows 16*gm..)
  const int nh = wv >> 3;  // GEMM1: s-half; GEMM2: d-half (256-wide)

  // ---- stage Q (f32 -> bf16) once ----
  const float4* Qb  = (const float4*)(Q   + (size_t)b * kH * kD);
  const float4* Qpb = (const float4*)(Qpe + (size_t)b * kH * kP);
  for (int u = 0; u < 32; ++u) {
    int v = u * kThreads + tid;                 // 16384 float4
    int g = v >> 7, d = (v & 127) << 2;
    float4 x = Qb[v];
    __bf16* q = &sQ[g * kDT + d];
    q[0] = (__bf16)x.x; q[1] = (__bf16)x.y; q[2] = (__bf16)x.z; q[3] = (__bf16)x.w;
  }
  for (int u = 0; u < 4; ++u) {
    int v = u * kThreads + tid;                 // 2048 float4
    int g = v >> 4, d = kD + ((v & 15) << 2);
    float4 x = Qpb[v];
    __bf16* q = &sQ[g * kDT + d];
    q[0] = (__bf16)x.x; q[1] = (__bf16)x.y; q[2] = (__bf16)x.z; q[3] = (__bf16)x.w;
  }
  if (tid < kH) { sM[tid] = -__builtin_inff(); sL[tid] = 0.0f; }

  v8f acc[16];  // 16x256 f32 out accumulator per wave (128 VGPRs)
#pragma unroll
  for (int j = 0; j < 16; ++j) acc[j] = (v8f){0,0,0,0,0,0,0,0};

  const int sbase0 = split * kChunk;
  for (int t = 0; t < kNT; ++t) {
    const int sb = sbase0 + t * kTile;
    __syncthreads();   // previous tile fully consumed

    // ---- stage K tile: f32 -> bf16, row-major + transposed copy ----
    const float4* Kb  = (const float4*)(K   + ((size_t)b * kS + sb) * kD);
    const float4* Kpb = (const float4*)(Kpe + ((size_t)b * kS + sb) * kP);
#pragma unroll
    for (int u = 0; u < 8; ++u) {
      int v = u * kThreads + tid;               // 4096 float4 = 32x512 f32
      int s = v >> 7, d = (v & 127) << 2;
      float4 x = Kb[v];
      __bf16 h0 = (__bf16)x.x, h1 = (__bf16)x.y, h2 = (__bf16)x.z, h3 = (__bf16)x.w;
      __bf16* kr = &sK[s * kDT + d];
      kr[0] = h0; kr[1] = h1; kr[2] = h2; kr[3] = h3;
      sKt[(d + 0) * kTile + s] = h0;
      sKt[(d + 1) * kTile + s] = h1;
      sKt[(d + 2) * kTile + s] = h2;
      sKt[(d + 3) * kTile + s] = h3;
    }
    {
      int v = tid;                              // 512 float4 = 32x64 f32
      int s = v >> 4, d = kD + ((v & 15) << 2);
      float4 x = Kpb[v];
      __bf16 h0 = (__bf16)x.x, h1 = (__bf16)x.y, h2 = (__bf16)x.z, h3 = (__bf16)x.w;
      __bf16* kr = &sK[s * kDT + d];
      kr[0] = h0; kr[1] = h1; kr[2] = h2; kr[3] = h3;
      sKt[(d + 0) * kTile + s] = h0;
      sKt[(d + 1) * kTile + s] = h1;
      sKt[(d + 2) * kTile + s] = h2;
      sKt[(d + 3) * kTile + s] = h3;
    }
    __syncthreads();

    // ---- GEMM1: scores[128 x 32] = Q(bf16) x K^T(bf16), f32 acc ----
    {
      v8f sc = (v8f){0,0,0,0,0,0,0,0};
      const __bf16* qrow = &sQ[(gm * 16 + col) * kDT];
      const __bf16* krow = &sK[(nh * 16 + col) * kDT + (lolane ? 0 : 16)];
      for (int ks = 0; ks < kDT / 32; ++ks) {   // 18 k-steps
        v16bf a  = frag_ilv(qrow + ks * 32, lolane);
        v16bf bb = frag_row(krow + ks * 32);
        sc = __builtin_amdgcn_wmma_f32_16x16x32_bf16(
            false, a, false, bb, (short)0, sc, false, false);
      }
#pragma unroll
      for (int i = 0; i < 8; ++i) {
        int g = gm * 16 + i + (lolane ? 0 : 8);
        sS[g * kTile + nh * 16 + col] = sc[i] * 0.0416666679f;  // 1/sqrt(576)
      }
    }
    __syncthreads();

    // ---- online softmax over this 32-key tile (one thread per head) ----
    if (tid < kH) {
      const float4* sp = (const float4*)&sS[tid * kTile];
      float vbuf[kTile];
      float mo = sM[tid];
      float mx = mo;
#pragma unroll
      for (int j = 0; j < kTile / 4; ++j) {
        float4 x = sp[j];
        vbuf[4 * j + 0] = x.x; vbuf[4 * j + 1] = x.y;
        vbuf[4 * j + 2] = x.z; vbuf[4 * j + 3] = x.w;
      }
#pragma unroll
      for (int j = 0; j < kTile; ++j) mx = fmaxf(mx, vbuf[j]);
      float alpha = __expf(mo - mx);
      float sum = 0.0f;
      __bf16* pr = &sP[tid * kTile];
#pragma unroll
      for (int j = 0; j < kTile; ++j) {
        float p = __expf(vbuf[j] - mx);
        sum += p;
        pr[j] = (__bf16)p;
      }
      sM[tid] = mx;
      sL[tid] = sL[tid] * alpha + sum;
      sA[tid] = alpha;
    }
    __syncthreads();

    // ---- rescale accumulators, then GEMM2: out += P(bf16) x K(bf16) ----
    {
      float av[8];
#pragma unroll
      for (int i = 0; i < 8; ++i)
        av[i] = sA[gm * 16 + i + (lolane ? 0 : 8)];
#pragma unroll
      for (int j = 0; j < 16; ++j)
#pragma unroll
        for (int i = 0; i < 8; ++i) acc[j][i] *= av[i];

      v16bf a2 = frag_ilv(&sP[(gm * 16 + col) * kTile], lolane);
#pragma unroll
      for (int j = 0; j < 16; ++j) {
        int n = nh * 256 + j * 16;
        v16bf b2 = frag_row(&sKt[(n + col) * kTile + (lolane ? 0 : 16)]);
        acc[j] = __builtin_amdgcn_wmma_f32_16x16x32_bf16(
            false, a2, false, b2, (short)0, acc[j], false, false);
      }
    }
  }

  // ---- write split partials (un-normalized out, plus m,l) ----
  float* pob = po + ((size_t)(b * kSplits + split) * kH) * kD;
#pragma unroll
  for (int j = 0; j < 16; ++j) {
#pragma unroll
    for (int i = 0; i < 8; ++i) {
      int g = gm * 16 + i + (lolane ? 0 : 8);
      int d = nh * 256 + j * 16 + col;
      pob[(size_t)g * kD + d] = acc[j][i];
    }
  }
  if (tid < kH) {
    int idx = (b * kSplits + split) * kH + tid;
    pm[idx] = sM[tid];
    pl[idx] = sL[tid];
  }
}

// ---- split combine: log-sum-exp merge of the 8 S-splits ----
__global__ __launch_bounds__(128) void mla_combine(
    const float* __restrict__ po, const float* __restrict__ pm,
    const float* __restrict__ pl, float* __restrict__ out)
{
  int bg = blockIdx.x;            // b*128 + g
  int b = bg >> 7, g = bg & 127;
  float m[kSplits], w[kSplits];
  float M = -__builtin_inff();
#pragma unroll
  for (int s = 0; s < kSplits; ++s) {
    m[s] = pm[(b * kSplits + s) * kH + g];
    M = fmaxf(M, m[s]);
  }
  float L = 0.0f;
#pragma unroll
  for (int s = 0; s < kSplits; ++s) {
    w[s] = __expf(m[s] - M);
    L += pl[(b * kSplits + s) * kH + g] * w[s];
  }
  float inv = 1.0f / L;
  int tid = threadIdx.x;          // 128 threads x float4 = 512 dims
  float4 a = {0.0f, 0.0f, 0.0f, 0.0f};
#pragma unroll
  for (int s = 0; s < kSplits; ++s) {
    const float4* p =
        (const float4*)(po + ((size_t)(b * kSplits + s) * kH + g) * kD);
    float4 x = p[tid];
    float ws = w[s] * inv;
    a.x += ws * x.x; a.y += ws * x.y; a.z += ws * x.z; a.w += ws * x.w;
  }
  ((float4*)(out + (size_t)bg * kD))[tid] = a;
}

extern "C" void kernel_launch(void* const* d_in, const int* in_sizes, int n_in,
                              void* d_out, int out_size, void* d_ws, size_t ws_size,
                              hipStream_t stream) {
  (void)in_sizes; (void)n_in; (void)out_size; (void)ws_size;
  const float* Q   = (const float*)d_in[0];
  const float* Qpe = (const float*)d_in[1];
  const float* K   = (const float*)d_in[2];
  const float* Kpe = (const float*)d_in[3];
  float* out = (float*)d_out;

  float* po = (float*)d_ws;                              // 32*8*128*512 f32 = 64 MB
  size_t poN = (size_t)kB * kSplits * kH * kD;
  float* pm = po + poN;                                  // 32*8*128 f32
  float* pl = pm + (size_t)kB * kSplits * kH;            // 32*8*128 f32

  mla_split<<<dim3(kSplits, kB), kThreads, 0, stream>>>(Q, Qpe, K, Kpe, po, pm, pl);
  mla_combine<<<dim3(kB * kH), 128, 0, stream>>>(po, pm, pl, out);
}
